// Transformer_70196945486070
// MI455X (gfx1250) — compile-verified, matrix operand order checked by reading, not run on hardware
//
#include <hip/hip_runtime.h>

typedef __attribute__((ext_vector_type(16))) __bf16 v16bf;
typedef __attribute__((ext_vector_type(8)))  __bf16 v8bf;
typedef __attribute__((ext_vector_type(8)))  float  v8f;

__device__ __forceinline__ v8f wmma_bf16(v16bf a, v16bf b, v8f c) {
  return __builtin_amdgcn_wmma_f32_16x16x32_bf16(false, a, false, b, (short)0, c, false, false);
}

// ---- fragment loaders (CDNA5 wave32 layouts, 16-bit 16x32 A / 32x16 B) ----
// per lane: idx = lane&15 selects row (A) / col (B); K pattern:
// lanes 0-15: K in {0..7}U{16..23}; lanes 16-31: K in {8..15}U{24..31}
// => two contiguous 16-byte runs per lane -> 2x b128 loads
__device__ __forceinline__ v16bf frag_kcontig(const __bf16* base, int ld) {
  int lane = threadIdx.x & 31;
  int idx  = lane & 15;
  int kb   = (lane & 16) ? 8 : 0;
  const __bf16* p = base + (size_t)idx * ld + kb;
  v8bf lo = *(const v8bf*)(p);
  v8bf hi = *(const v8bf*)(p + 16);
  v16bf f;
#pragma unroll
  for (int i = 0; i < 8; ++i) { f[i] = lo[i]; f[8+i] = hi[i]; }
  return f;
}
// B from f32 weight matrix W[N,K] row-major (convert in regs; used only on tiny paths)
__device__ __forceinline__ v16bf frag_b_w32(const float* W, int ldw) {
  int lane = threadIdx.x & 31;
  int nn   = lane & 15;
  int kb   = (lane & 16) ? 8 : 0;
  const float* p = W + (size_t)nn * ldw + kb;
  v16bf f;
#pragma unroll
  for (int j = 0; j < 4; ++j) {
    f[2*j]     = (__bf16)p[2*j];
    f[2*j+1]   = (__bf16)p[2*j+1];
    f[8+2*j]   = (__bf16)p[16+2*j];
    f[8+2*j+1] = (__bf16)p[16+2*j+1];
  }
  return f;
}
// one 16x16 output tile: A[16,K] bf16 (ld lda) x W[N,K] f32 (row n0 at W)
__device__ __forceinline__ v8f tile16(const __bf16* A, int lda, const float* W, int ldw, int K) {
  v8f c = {};
  for (int k0 = 0; k0 < K; k0 += 32)
    c = wmma_bf16(frag_kcontig(A + k0, lda), frag_b_w32(W + k0, ldw), c);
  return c;
}

// ---------------- weight convert: f32 [2 x nper] (d-stride sstride) -> bf16 packed ----------------
__global__ __launch_bounds__(256) void cvtw_k(const float* __restrict__ src, long sstride,
                                              int nper, __bf16* __restrict__ dst) {
  long i = ((long)blockIdx.x * 256 + threadIdx.x) * 4;
  int d = (int)(i >= nper);
  long j = i - (long)d * nper;
  const float4 v = *(const float4*)(src + (long)d * sstride + j);
  __bf16* o = dst + i;
  o[0] = (__bf16)v.x; o[1] = (__bf16)v.y; o[2] = (__bf16)v.z; o[3] = (__bf16)v.w;
}

// ---------------- K0: embedding + positional encoding -> Y f32 ----------------
__global__ __launch_bounds__(256) void embed_k(const float* __restrict__ t1,
                                               const float* __restrict__ t2,
                                               const float* __restrict__ Wemb,
                                               float* __restrict__ Y) {
  long idx = (long)blockIdx.x * 256 + threadIdx.x;   // [2,16,512,512]
  int j = idx & 511;
  int s = (idx >> 9) & 511;
  int b = (idx >> 18) & 15;
  int d = (int)(idx >> 22) & 1;
  const float* t = d ? t2 : t1;
  float t0 = t[((size_t)b * 512 + s) * 2 + 0];
  float t1v = t[((size_t)b * 512 + s) * 2 + 1];
  float val = (t0 * Wemb[2*j] + t1v * Wemb[2*j+1]) * 22.62741699796952f; // sqrt(512)
  int i = j >> 1;
  float div = __expf(-9.210340371976184f * (float)(2*i) * (1.0f/512.0f));
  float arg = (float)s * div;
  val += (j & 1) ? __cosf(arg) : __sinf(arg);
  Y[idx] = val;
}

// ---------------- RMSNorm helper for M=16 block kernels ----------------
__device__ void rms_block16(const float* xs, const float* w, __bf16* xn) {
  int wv = threadIdx.x >> 5, lane = threadIdx.x & 31;
#pragma unroll
  for (int rr = 0; rr < 2; ++rr) {
    int row = wv * 2 + rr;
    const float* x = xs + (size_t)row * 512;
    float ss = 0.f;
    for (int i = 0; i < 16; ++i) { float v = x[lane + 32*i]; ss += v*v; }
    for (int m = 16; m; m >>= 1) ss += __shfl_xor(ss, m, 32);
    float r = rsqrtf(ss * (1.f/512.f) + 1e-5f);
    __bf16* xo = xn + (size_t)row * 512;
    for (int i = 0; i < 16; ++i) { int j = lane + 32*i; xo[j] = (__bf16)(w[j] * x[j] * r); }
  }
}

// ---------------- K1: encoder (seq len 1 => attn == Wv,Wo path) ----------------
__global__ __launch_bounds__(256) void encoder_k(const float* __restrict__ src,
                                                 const float* __restrict__ ewv,
                                                 const float* __restrict__ ewo,
                                                 const float* __restrict__ ew1,
                                                 const float* __restrict__ ew2,
                                                 const float* __restrict__ ew3,
                                                 const float* __restrict__ en1,
                                                 const float* __restrict__ en2,
                                                 float* __restrict__ ENC,
                                                 __bf16* __restrict__ ENCB,
                                                 __bf16* __restrict__ encV,
                                                 __bf16* __restrict__ encG) {
  __shared__ float xs[16][512];
  __shared__ alignas(16) __bf16 xn[16][512];
  int t = threadIdx.x, wv = t >> 5, lane = t & 31;
  int n = lane & 15, m0 = (lane & 16) ? 8 : 0;
  for (int i = t; i < 16*512; i += 256) {
    int b = i >> 9, j = i & 511;
    xs[b][j] = src[(size_t)b*512 + j] + ((j & 1) ? 1.f : 0.f);  // pe(pos=0)
  }
  __syncthreads();
  for (int l = 0; l < 6; ++l) {
    const float* wvp = ewv + (size_t)l * 262144;
    const float* wop = ewo + (size_t)l * 262144;
    const float* w1p = ew1 + (size_t)l * 1048576;
    const float* w2p = ew2 + (size_t)l * 1048576;
    const float* w3p = ew3 + (size_t)l * 1048576;
    rms_block16(&xs[0][0], en1 + l*512, &xn[0][0]);
    __syncthreads();
    // v = xn @ wv^T  -> encV
    for (int nt = wv; nt < 32; nt += 8) {
      v8f c = tile16(&xn[0][0], 512, wvp + (size_t)nt*16*512, 512, 512);
#pragma unroll
      for (int r = 0; r < 8; ++r) encV[(size_t)(m0+r)*512 + nt*16 + n] = (__bf16)c[r];
    }
    __threadfence_block(); __syncthreads();
    // x += v @ wo^T
    for (int nt = wv; nt < 32; nt += 8) {
      v8f c = tile16(encV, 512, wop + (size_t)nt*16*512, 512, 512);
#pragma unroll
      for (int r = 0; r < 8; ++r) xs[m0+r][nt*16 + n] += c[r];
    }
    __syncthreads();
    rms_block16(&xs[0][0], en2 + l*512, &xn[0][0]);
    __syncthreads();
    // g = silu(xn@w1^T) * (xn@w3^T)  -> encG [16,2048]
    for (int nt = wv; nt < 128; nt += 8) {
      v8f c1 = tile16(&xn[0][0], 512, w1p + (size_t)nt*16*512, 512, 512);
      v8f c3 = tile16(&xn[0][0], 512, w3p + (size_t)nt*16*512, 512, 512);
#pragma unroll
      for (int r = 0; r < 8; ++r) {
        float x1 = c1[r];
        float g  = (x1 / (1.f + __expf(-x1))) * c3[r];
        encG[(size_t)(m0+r)*2048 + nt*16 + n] = (__bf16)g;
      }
    }
    __threadfence_block(); __syncthreads();
    // x += g @ w2^T   (K=2048)
    for (int nt = wv; nt < 32; nt += 8) {
      v8f c = tile16(encG, 2048, w2p + (size_t)nt*16*2048, 2048, 2048);
#pragma unroll
      for (int r = 0; r < 8; ++r) xs[m0+r][nt*16 + n] += c[r];
    }
    __syncthreads();
  }
  for (int i = t; i < 16*512; i += 256) {
    float v = xs[i >> 9][i & 511];
    ENC[i]  = v;
    ENCB[i] = (__bf16)v;
  }
}

// ---------------- K2: cross-attn constants  CC[d,l,b,:] = (enc@wv_c^T)@wo_c^T ----------------
__global__ __launch_bounds__(256) void crossconst_k(const __bf16* __restrict__ ENCB,
                                                    const float* __restrict__ wvc,
                                                    const float* __restrict__ woc,
                                                    float* __restrict__ CC) {
  __shared__ alignas(16) __bf16 tmp[16][512];
  int dl = blockIdx.x;                         // d*6+l
  const float* Wv = wvc + (size_t)dl * 262144;
  const float* Wo = woc + (size_t)dl * 262144;
  int wv = threadIdx.x >> 5, lane = threadIdx.x & 31;
  int n = lane & 15, m0 = (lane & 16) ? 8 : 0;
  for (int nt = wv; nt < 32; nt += 8) {
    v8f c = tile16(ENCB, 512, Wv + (size_t)nt*16*512, 512, 512);
#pragma unroll
    for (int r = 0; r < 8; ++r) tmp[m0+r][nt*16 + n] = (__bf16)c[r];
  }
  __syncthreads();
  float* out = CC + (size_t)dl * 8192;
  for (int nt = wv; nt < 32; nt += 8) {
    v8f c = tile16(&tmp[0][0], 512, Wo + (size_t)nt*16*512, 512, 512);
#pragma unroll
    for (int r = 0; r < 8; ++r) out[(size_t)(m0+r)*512 + nt*16 + n] = c[r];
  }
}

// ---------------- K3: RMSNorm rows of Y -> bf16 ----------------
__global__ __launch_bounds__(256) void rmsnorm_k(const float* __restrict__ Y,
                                                 const float* __restrict__ wbase,
                                                 int wstride_d,
                                                 __bf16* __restrict__ out) {
  int wv = threadIdx.x >> 5, lane = threadIdx.x & 31;
  long row = (long)blockIdx.x * 8 + wv;        // 16384 rows
  int d = (int)(row >> 13);
  const float* w = wbase + (size_t)d * wstride_d;
  const float* x = Y + row * 512;
  float ss = 0.f;
  for (int i = 0; i < 16; ++i) { float v = x[lane + 32*i]; ss += v*v; }
  for (int m = 16; m; m >>= 1) ss += __shfl_xor(ss, m, 32);
  float r = rsqrtf(ss * (1.f/512.f) + 1e-5f);
  __bf16* o = out + row * 512;
  for (int i = 0; i < 16; ++i) { int j = lane + 32*i; o[j] = (__bf16)(w[j] * x[j] * r); }
}

// ---------------- K4: generic GEMM  C = A_bf16 @ W_bf16^T ----------------
// EPI 0: store bf16*scale to outB (col offset ocol)    EPI 1: Y += C (+ cross const)
template <int EPI>
__global__ __launch_bounds__(256) void gemm_k(const __bf16* __restrict__ A, int lda,
                                              const __bf16* __restrict__ Wbase,
                                              long wdstride, int ldw, int K,
                                              __bf16* __restrict__ outB, int ldo, int ocol,
                                              float scale,
                                              float* __restrict__ Y,
                                              const float* __restrict__ ccL) {
  int z = blockIdx.z;                          // d*16+b
  int d = z >> 4;
  int wvid = threadIdx.x >> 5;
  int mw = wvid & 1, nw = wvid >> 1;           // 2 x 4 waves -> 64 x 128 block tile
  long mbase = (long)z * 512 + blockIdx.y * 64 + mw * 32;
  int  nbase = blockIdx.x * 128 + nw * 32;
  const __bf16* W = Wbase + (size_t)d * wdstride;
  v8f acc[2][2] = {};
  for (int k0 = 0; k0 < K; k0 += 32) {
    v16bf a0 = frag_kcontig(A + (size_t)mbase * lda + k0, lda);
    v16bf a1 = frag_kcontig(A + (size_t)(mbase + 16) * lda + k0, lda);
    v16bf b0 = frag_kcontig(W + (size_t)nbase * ldw + k0, ldw);
    v16bf b1 = frag_kcontig(W + (size_t)(nbase + 16) * ldw + k0, ldw);
    acc[0][0] = wmma_bf16(a0, b0, acc[0][0]);
    acc[0][1] = wmma_bf16(a0, b1, acc[0][1]);
    acc[1][0] = wmma_bf16(a1, b0, acc[1][0]);
    acc[1][1] = wmma_bf16(a1, b1, acc[1][1]);
  }
  int lane = threadIdx.x & 31, n = lane & 15, m0 = (lane & 16) ? 8 : 0;
#pragma unroll
  for (int mi = 0; mi < 2; ++mi)
#pragma unroll
    for (int nj = 0; nj < 2; ++nj)
#pragma unroll
      for (int r = 0; r < 8; ++r) {
        long row = mbase + 16*mi + m0 + r;
        int  col = nbase + 16*nj + n;
        if (EPI == 0) {
          outB[(size_t)row * ldo + ocol + col] = (__bf16)(acc[mi][nj][r] * scale);
        } else {
          float add = acc[mi][nj][r];
          if (ccL) add += ccL[(size_t)d * 49152 + (size_t)(z & 15) * 512 + col];
          Y[(size_t)row * 512 + col] += add;
        }
      }
}

// ---------------- K5: fused SwiGLU gate  G = silu(A@w1^T)*(A@w3^T) ----------------
__global__ __launch_bounds__(256) void ffn_gate_k(const __bf16* __restrict__ A,
                                                  const __bf16* __restrict__ W1b,
                                                  const __bf16* __restrict__ W3b,
                                                  long wdstride,
                                                  __bf16* __restrict__ G) {
  int z = blockIdx.z, d = z >> 4;
  int wvid = threadIdx.x >> 5;
  int mw = wvid & 1, nw = wvid >> 1;
  long mbase = (long)z * 512 + blockIdx.y * 64 + mw * 32;
  int  nbase = blockIdx.x * 128 + nw * 32;
  const __bf16* W1 = W1b + (size_t)d * wdstride;
  const __bf16* W3 = W3b + (size_t)d * wdstride;
  v8f c1[2][2] = {}, c3[2][2] = {};
  for (int k0 = 0; k0 < 512; k0 += 32) {
    v16bf a0 = frag_kcontig(A + (size_t)mbase * 512 + k0, 512);
    v16bf a1 = frag_kcontig(A + (size_t)(mbase + 16) * 512 + k0, 512);
    v16bf b10 = frag_kcontig(W1 + (size_t)nbase * 512 + k0, 512);
    v16bf b11 = frag_kcontig(W1 + (size_t)(nbase + 16) * 512 + k0, 512);
    v16bf b30 = frag_kcontig(W3 + (size_t)nbase * 512 + k0, 512);
    v16bf b31 = frag_kcontig(W3 + (size_t)(nbase + 16) * 512 + k0, 512);
    c1[0][0] = wmma_bf16(a0, b10, c1[0][0]); c1[0][1] = wmma_bf16(a0, b11, c1[0][1]);
    c1[1][0] = wmma_bf16(a1, b10, c1[1][0]); c1[1][1] = wmma_bf16(a1, b11, c1[1][1]);
    c3[0][0] = wmma_bf16(a0, b30, c3[0][0]); c3[0][1] = wmma_bf16(a0, b31, c3[0][1]);
    c3[1][0] = wmma_bf16(a1, b30, c3[1][0]); c3[1][1] = wmma_bf16(a1, b31, c3[1][1]);
  }
  int lane = threadIdx.x & 31, n = lane & 15, m0 = (lane & 16) ? 8 : 0;
#pragma unroll
  for (int mi = 0; mi < 2; ++mi)
#pragma unroll
    for (int nj = 0; nj < 2; ++nj)
#pragma unroll
      for (int r = 0; r < 8; ++r) {
        long row = mbase + 16*mi + m0 + r;
        int  col = nbase + 16*nj + n;
        float x1 = c1[mi][nj][r];
        float g  = (x1 / (1.f + __expf(-x1))) * c3[mi][nj][r];
        G[(size_t)row * 2048 + col] = (__bf16)g;
      }
}

// ---------------- V transpose: VT[z,h,n,k] = V[z,k,h*64+n] ----------------
__global__ __launch_bounds__(256) void vtrans_k(const __bf16* __restrict__ QKV,
                                                __bf16* __restrict__ VT) {
  long i = (long)blockIdx.x * 256 + threadIdx.x;   // 2*16*8*64*512
  int k = (int)(i & 511);
  int n = (int)((i >> 9) & 63);
  int h = (int)((i >> 15) & 7);
  int z = (int)(i >> 18);
  VT[i] = QKV[((size_t)z * 512 + k) * 1536 + 1024 + h * 64 + n];
}

// ---------------- K6: causal flash attention (dk=64), wave tile 16q x 64k ----------------
__global__ __launch_bounds__(128) void attn_k(const __bf16* __restrict__ QKV,
                                              const __bf16* __restrict__ VT,
                                              __bf16* __restrict__ O) {
  __shared__ alignas(16) __bf16 pbuf[4][16][72];
  int z = blockIdx.z, h = blockIdx.y;
  int wvid = threadIdx.x >> 5, lane = threadIdx.x & 31;
  int n = lane & 15, m0 = (lane & 16) ? 8 : 0;
  const __bf16* base = QKV + (size_t)z * 512 * 1536;
  const __bf16* Qp = base + h * 64;            // q pre-scaled by 1/8
  const __bf16* Kp = base + 512 + h * 64;
  const __bf16* VTz = VT + ((size_t)z * 8 + h) * 64 * 512;  // [64][512]
  int q0 = blockIdx.x * 64 + wvid * 16;
  v16bf aq[2];
#pragma unroll
  for (int kk = 0; kk < 2; ++kk)
    aq[kk] = frag_kcontig(Qp + (size_t)q0 * 1536 + 32*kk, 1536);
  float rmax[8], rsum[8];
  v8f o4[4] = {};
#pragma unroll
  for (int r = 0; r < 8; ++r) { rmax[r] = -1e30f; rsum[r] = 0.f; }
  for (int kb = 0; kb <= q0 + 15; kb += 64) {
    // scores S[16,64] = Q Kt
    v8f s[4];
#pragma unroll
    for (int nj = 0; nj < 4; ++nj) {
      v16bf bk0 = frag_kcontig(Kp + (size_t)(kb + 16*nj) * 1536, 1536);
      v16bf bk1 = frag_kcontig(Kp + (size_t)(kb + 16*nj) * 1536 + 32, 1536);
      v8f acc = {};
      acc = wmma_bf16(aq[0], bk0, acc);
      acc = wmma_bf16(aq[1], bk1, acc);
      s[nj] = acc;
    }
    // causal mask + block row max
    float mb[8];
#pragma unroll
    for (int r = 0; r < 8; ++r) mb[r] = -1e30f;
#pragma unroll
    for (int nj = 0; nj < 4; ++nj)
#pragma unroll
      for (int r = 0; r < 8; ++r) {
        int row = q0 + m0 + r;
        int col = kb + 16*nj + n;
        float v = s[nj][r];
        if (col > row) v = -1e30f;
        s[nj][r] = v;
        mb[r] = fmaxf(mb[r], v);
      }
#pragma unroll
    for (int r = 0; r < 8; ++r)
      for (int m = 8; m; m >>= 1) mb[r] = fmaxf(mb[r], __shfl_xor(mb[r], m, 32));
    // online softmax update
    float scl[8];
#pragma unroll
    for (int r = 0; r < 8; ++r) {
      float nm = fmaxf(rmax[r], mb[r]);
      scl[r] = __expf(rmax[r] - nm);
      rmax[r] = nm;
    }
    float ps[8];
#pragma unroll
    for (int r = 0; r < 8; ++r) ps[r] = 0.f;
#pragma unroll
    for (int nj = 0; nj < 4; ++nj)
#pragma unroll
      for (int r = 0; r < 8; ++r) {
        float p = __expf(s[nj][r] - rmax[r]);
        s[nj][r] = p;
        ps[r] += p;
      }
#pragma unroll
    for (int r = 0; r < 8; ++r)
      for (int m = 8; m; m >>= 1) ps[r] += __shfl_xor(ps[r], m, 32);
#pragma unroll
    for (int r = 0; r < 8; ++r) rsum[r] = rsum[r] * scl[r] + ps[r];
#pragma unroll
    for (int nj = 0; nj < 4; ++nj)
#pragma unroll
      for (int r = 0; r < 8; ++r) o4[nj][r] *= scl[r];
    // P tile -> LDS (C layout -> A layout)
#pragma unroll
    for (int nj = 0; nj < 4; ++nj)
#pragma unroll
      for (int r = 0; r < 8; ++r)
        pbuf[wvid][m0 + r][16*nj + n] = (__bf16)s[nj][r];
    asm volatile("s_wait_dscnt 0x0" ::: "memory");
    // O += P @ V   (B frags from transposed V: k-contiguous)
#pragma unroll
    for (int kk = 0; kk < 2; ++kk) {
      v16bf ap = frag_kcontig(&pbuf[wvid][0][32*kk], 72);
#pragma unroll
      for (int nj2 = 0; nj2 < 4; ++nj2) {
        v16bf bv = frag_kcontig(VTz + (size_t)(16*nj2) * 512 + kb + 32*kk, 512);
        o4[nj2] = wmma_bf16(ap, bv, o4[nj2]);
      }
    }
    asm volatile("s_wait_dscnt 0x0" ::: "memory");
  }
  // normalize + store to O (attn output buffer, [rows,512], head col offset)
#pragma unroll
  for (int nj2 = 0; nj2 < 4; ++nj2)
#pragma unroll
    for (int r = 0; r < 8; ++r) {
      int row = q0 + m0 + r;
      float v = o4[nj2][r] / rsum[r];
      O[((size_t)z * 512 + row) * 512 + h * 64 + 16*nj2 + n] = (__bf16)v;
    }
}

// ---------------- K7: final projection  out = Y @ W_proj^T ----------------
__global__ __launch_bounds__(256) void finalproj_k(const float* __restrict__ Y,
                                                   const float* __restrict__ Wp,
                                                   float* __restrict__ out) {
  int wvid = threadIdx.x >> 5, lane = threadIdx.x & 31;
  long row = (long)blockIdx.x * 8 + wvid;
  const float* x = Y + row * 512;
  float s0 = 0.f, s1 = 0.f;
  for (int i = 0; i < 16; ++i) {
    int j = lane + 32*i;
    float v = x[j];
    s0 += v * Wp[j];
    s1 += v * Wp[512 + j];
  }
  for (int m = 16; m; m >>= 1) { s0 += __shfl_xor(s0, m, 32); s1 += __shfl_xor(s1, m, 32); }
  if (lane == 0) {
    int s = (int)(row & 511), b = (int)((row >> 9) & 15), d = (int)(row >> 13);
    size_t oi = ((size_t)b * 1024 + (size_t)d * 512 + s) * 2;
    out[oi]     = s0;
    out[oi + 1] = s1;
  }
}

extern "C" void kernel_launch(void* const* d_in, const int* in_sizes, int n_in,
                              void* d_out, int out_size, void* d_ws, size_t ws_size,
                              hipStream_t stream) {
  (void)in_sizes; (void)n_in; (void)out_size; (void)ws_size;
  const float* src    = (const float*)d_in[0];
  const float* tgt1   = (const float*)d_in[1];
  const float* tgt2   = (const float*)d_in[2];
  const float* W_emb  = (const float*)d_in[5];
  const float* W_proj = (const float*)d_in[6];
  const float* e_wv   = (const float*)d_in[9];
  const float* e_wo   = (const float*)d_in[10];
  const float* e_w1   = (const float*)d_in[11];
  const float* e_w2   = (const float*)d_in[12];
  const float* e_w3   = (const float*)d_in[13];
  const float* e_n1   = (const float*)d_in[14];
  const float* e_n2   = (const float*)d_in[15];
  const float* w_q    = (const float*)d_in[16];
  const float* w_k    = (const float*)d_in[17];
  const float* w_v    = (const float*)d_in[18];
  const float* w_o    = (const float*)d_in[19];
  const float* w_vc   = (const float*)d_in[22];
  const float* w_oc   = (const float*)d_in[23];
  const float* w_1    = (const float*)d_in[24];
  const float* w_2    = (const float*)d_in[25];
  const float* w_3    = (const float*)d_in[26];
  const float* n_1    = (const float*)d_in[27];
  const float* n_3    = (const float*)d_in[29];
  float* out = (float*)d_out;

  // ---- workspace layout (bytes) ----
  char* ws = (char*)d_ws;
  float*  Y    = (float*)ws;                                   // 32 MB
  __bf16* Abuf = (__bf16*)(ws + (size_t)33554432);             // 16 MB
  __bf16* QKV  = (__bf16*)(ws + (size_t)50331648);             // 48 MB (in 64MB region)
  __bf16* G    = (__bf16*)(ws + (size_t)50331648);             // 64 MB (reused in FFN phase)
  __bf16* VT   = (__bf16*)(ws + (size_t)100663296);            // 16 MB (tail of G region, attn phase)
  char* tail   = ws + (size_t)117440512;
  float*  ENC  = (float*)tail;                                 // 32 KB
  __bf16* ENCB = (__bf16*)(tail + 32768);                      // 16 KB
  float*  CC   = (float*)(tail + 49152);                       // 384 KB
  __bf16* encV = (__bf16*)(tail + 49152 + 393216);             // 16 KB
  __bf16* encG = (__bf16*)(tail + 49152 + 393216 + 16384);     // 64 KB
  __bf16* WL   = (__bf16*)(ws + (size_t)117964800);            // 16 MB per-layer bf16 weights

  // per-layer bf16 weight arena offsets (elements)
  __bf16* WLq = WL;                    // [2][512][512]
  __bf16* WLk = WL + 524288;
  __bf16* WLv = WL + 1048576;
  __bf16* WLo = WL + 1572864;
  __bf16* WL1 = WL + 2097152;          // [2][2048][512]
  __bf16* WL3 = WL + 4194304;
  __bf16* WL2 = WL + 6291456;          // [2][512][2048]

  // embeddings + PE
  embed_k<<<dim3(65536), 256, 0, stream>>>(tgt1, tgt2, W_emb, Y);
  // encoder + cross-attention constants
  encoder_k<<<dim3(1), 256, 0, stream>>>(src, e_wv, e_wo, e_w1, e_w2, e_w3,
                                         e_n1, e_n2, ENC, ENCB, encV, encG);
  crossconst_k<<<dim3(12), 256, 0, stream>>>(ENCB, w_vc, w_oc, CC);

  for (int l = 0; l < 6; ++l) {
    size_t o512  = (size_t)l * 512 * 512;
    size_t o2048 = (size_t)l * 2048 * 512;
    // stage this layer's decoder weights as bf16
    cvtw_k<<<dim3(512),  256, 0, stream>>>(w_q + o512,  1572864, 262144,  WLq);
    cvtw_k<<<dim3(512),  256, 0, stream>>>(w_k + o512,  1572864, 262144,  WLk);
    cvtw_k<<<dim3(512),  256, 0, stream>>>(w_v + o512,  1572864, 262144,  WLv);
    cvtw_k<<<dim3(512),  256, 0, stream>>>(w_o + o512,  1572864, 262144,  WLo);
    cvtw_k<<<dim3(2048), 256, 0, stream>>>(w_1 + o2048, 6291456, 1048576, WL1);
    cvtw_k<<<dim3(2048), 256, 0, stream>>>(w_3 + o2048, 6291456, 1048576, WL3);
    cvtw_k<<<dim3(2048), 256, 0, stream>>>(w_2 + o2048, 6291456, 1048576, WL2);
    // yn = rms(y, n1)
    rmsnorm_k<<<dim3(2048), 256, 0, stream>>>(Y, n_1 + l*512, 6*512, Abuf);
    // QKV projections (q scaled by 1/sqrt(64))
    gemm_k<0><<<dim3(4, 8, 32), 256, 0, stream>>>(Abuf, 512, WLq, 262144, 512, 512,
                                                  QKV, 1536, 0,    0.125f, nullptr, nullptr);
    gemm_k<0><<<dim3(4, 8, 32), 256, 0, stream>>>(Abuf, 512, WLk, 262144, 512, 512,
                                                  QKV, 1536, 512,  1.0f,   nullptr, nullptr);
    gemm_k<0><<<dim3(4, 8, 32), 256, 0, stream>>>(Abuf, 512, WLv, 262144, 512, 512,
                                                  QKV, 1536, 1024, 1.0f,   nullptr, nullptr);
    // transpose V per head for contiguous PV fragments
    vtrans_k<<<dim3(32768), 256, 0, stream>>>(QKV, VT);
    // causal flash attention -> Abuf
    attn_k<<<dim3(8, 8, 32), 128, 0, stream>>>(QKV, VT, Abuf);
    // y += attnout @ wo^T + cross_const
    gemm_k<1><<<dim3(4, 8, 32), 256, 0, stream>>>(Abuf, 512, WLo, 262144, 512, 512,
                                                  nullptr, 0, 0, 1.0f, Y, CC + (size_t)l*8192);
    // yn = rms(y, n3)
    rmsnorm_k<<<dim3(2048), 256, 0, stream>>>(Y, n_3 + l*512, 6*512, Abuf);
    // g = silu(yn@w1^T) * (yn@w3^T)
    ffn_gate_k<<<dim3(16, 8, 32), 256, 0, stream>>>(Abuf, WL1, WL3, 1048576, G);
    // y += g @ w2^T
    gemm_k<1><<<dim3(4, 8, 32), 256, 0, stream>>>(G, 2048, WL2, 1048576, 2048, 2048,
                                                  nullptr, 0, 0, 1.0f, Y, nullptr);
  }
  // final projection
  finalproj_k<<<dim3(2048), 256, 0, stream>>>(Y, W_proj, out);
}
